// MP_GNN_30580167147633
// MI455X (gfx1250) — matrix operand order repaired and checked by reading, not compiled, not run on hardware
//
#include <hip/hip_runtime.h>

typedef __attribute__((ext_vector_type(16))) _Float16 v16h;
typedef __attribute__((ext_vector_type(8)))  _Float16 v8h;
typedef __attribute__((ext_vector_type(2)))  _Float16 v2h;
typedef __attribute__((ext_vector_type(8)))  float    v8f;

#define NNODES     50000
#define NEDGES_RAW 800000
#define NEDGES_TOT 850000
#define EMB        64
#define TPW        4        // tiles per wave (grid shrink factor)

__device__ __forceinline__ void wave_sync() { __builtin_amdgcn_wave_barrier(); }

// ---------------- WMMA fragment helpers (ISA 7.12.2 layouts, wave32) ----------

// A-matrix 16x32 f16 fragment from an LDS tile (row-major, `stride` halfs/row).
// Lane layout: m = lane&15, h = lane>>4. Halves 0..7 = K(kt*32 + 8h .. +7),
// halves 8..15 = K(kt*32 + 16 + 8h .. +7)  -> two contiguous 16B LDS loads.
__device__ __forceinline__ v16h frag_a(const _Float16* tile, int lane, int kt, int stride) {
  const int m = lane & 15, h = lane >> 4;
  const _Float16* p = tile + m * stride + kt * 32 + h * 8;
  const v8h lo = *(const v8h*)p;
  const v8h hi = *(const v8h*)(p + 16);
  return __builtin_shufflevector(lo, hi, 0, 1, 2, 3, 4, 5, 6, 7,
                                 8, 9, 10, 11, 12, 13, 14, 15);
}

// B-matrix 32x16 f16 fragment from LDS weights pre-swizzled into fragment
// order: dst[((kt*4 + nt)*32 + lane)*16 + e]  -> one aligned 32B LDS load.
__device__ __forceinline__ v16h frag_b(const _Float16* wsw, int lane, int kt, int nt) {
  return *(const v16h*)(wsw + (((kt << 2) + nt) * 32 + lane) * 16);
}

// Stage a K x 64 f32 weight matrix into LDS as f16 in B-fragment order,
// zero-padding K rows beyond DIN. KT = padded-K/32.
template <int KT, int DIN>
__device__ __forceinline__ void stage_w_sw(_Float16* dst, const float* __restrict__ w,
                                           int tid) {
  for (int c = tid; c < KT * 4 * 32; c += 256) {
    const int ln = c & 31, nt = (c >> 5) & 3, kt = c >> 7;
    const int n  = (ln & 15) + nt * 16;
    const int kb = kt * 32 + ((ln >> 4) << 3);
    v16h t;
#pragma unroll
    for (int e = 0; e < 16; ++e) {
      const int k = kb + (e & 7) + ((e >> 3) << 4);
      t[e] = (_Float16)((k < DIN) ? w[k * EMB + n] : 0.0f);
    }
    *(v16h*)(dst + c * 16) = t;
  }
}

// 16xK @ Kx64 -> 16x64 accumulated into acc[4] (four 16-wide N tiles).
template <int KT>
__device__ __forceinline__ void gemm16(const _Float16* atile, int stride,
                                       const _Float16* wsw, int lane, v8f* acc) {
#pragma unroll
  for (int kt = 0; kt < KT; ++kt) {
    const v16h a = frag_a(atile, lane, kt, stride);
#pragma unroll
    for (int nt = 0; nt < 4; ++nt) {
      const v16h b = frag_b(wsw, lane, kt, nt);
      acc[nt] = __builtin_amdgcn_wmma_f32_16x16x32_f16(
          false, a, false, b, (short)0, acc[nt], false, false);
    }
  }
}

// ---------------- zero kernel (aggr init, graph-capture safe) -----------------
__global__ void gnn_zero_kernel(float* __restrict__ p, long n) {
  long i = (long)blockIdx.x * blockDim.x + threadIdx.x;
  const long st = (long)gridDim.x * blockDim.x;
  for (; i < n; i += st) p[i] = 0.0f;
}

// ---------------- per-edge: edge-MLP + msg-MLP + scatter-add ------------------
template <int ND, int ED, bool FIRST>
__global__ void __launch_bounds__(256) gnn_edge_msg_kernel(
    const float* __restrict__ x, const float* __restrict__ ea_in,
    const int* __restrict__ eidx,
    const float* __restrict__ w1e, const float* __restrict__ b1e,
    const float* __restrict__ w2e, const float* __restrict__ b2e,
    const float* __restrict__ w1m, const float* __restrict__ b1m,
    const float* __restrict__ w2m, const float* __restrict__ b2m,
    float* __restrict__ ea_out, float* __restrict__ aggr) {
  constexpr int TW   = 2 * ND + EMB;          // msg-MLP input width (== its K)
  constexpr int DINE = 2 * ND + ED;           // edge-MLP input width
  constexpr int KPE  = (DINE + 31) & ~31;     // padded to WMMA K granularity
  constexpr int KTE  = KPE / 32;
  constexpr int KTM  = TW / 32;

  extern __shared__ char smem[] __attribute__((aligned(32)));
  _Float16* w1e_s = (_Float16*)smem;                   // KPE*64 (swizzled)
  _Float16* w2e_s = w1e_s + KPE * EMB;                 // 64*64  (swizzled)
  _Float16* w1m_s = w2e_s + EMB * EMB;                 // TW*64  (swizzled)
  _Float16* w2m_s = w1m_s + TW * EMB;                  // 64*64  (swizzled)
  float*    bia_s = (float*)(w2m_s + EMB * EMB);       // 4*64 f32
  _Float16* tiles = (_Float16*)(bia_s + 4 * EMB);      // 8 waves * 16*TW
  _Float16* hids  = tiles + 8 * 16 * TW;               // 8 waves * 16*64
  int*      dids  = (int*)(hids + 8 * 16 * EMB);       // 8 waves * 16

  const int tid = threadIdx.x;
  stage_w_sw<KTE, DINE>(w1e_s, w1e, tid);
  stage_w_sw<2, EMB>(w2e_s, w2e, tid);
  stage_w_sw<KTM, TW>(w1m_s, w1m, tid);
  stage_w_sw<2, EMB>(w2m_s, w2m, tid);
  if (tid < EMB) {
    bia_s[tid]           = b1e[tid];
    bia_s[EMB + tid]     = b2e[tid];
    bia_s[2 * EMB + tid] = b1m[tid];
    bia_s[3 * EMB + tid] = b2m[tid];
  }
  __syncthreads();

  const int wave = tid >> 5, lane = tid & 31;
  const int lm = lane & 15, lh = lane >> 4;
  constexpr int NTILES = NEDGES_TOT / 16;  // 53125, exact

  _Float16* at = tiles + wave * 16 * TW;
  _Float16* ht = hids + wave * 16 * EMB;
  int*      dd = dids + wave * 16;

  for (int tile = blockIdx.x * 8 + wave; tile < NTILES; tile += (int)gridDim.x * 8) {
    const int ebase = tile * 16;

    // ---- gather 16 edges: [xi | xj | ea(+pad)] as f16; float2 loads, b32 LDS
    for (int r = 0; r < 16; ++r) {
      const int e = ebase + r;
      int s, d;
      if (e < NEDGES_RAW) { s = eidx[e]; d = eidx[NEDGES_RAW + e]; }
      else                { s = e - NEDGES_RAW; d = s; }      // self loop
      if (lane == 0) dd[r] = d;
      for (int c = lane * 2; c < ND; c += 64) {
        const float2 v = *(const float2*)(x + (long)d * ND + c);   // xi = x[dst]
        v2h hh = {(_Float16)v.x, (_Float16)v.y};
        *(v2h*)(at + r * TW + c) = hh;
      }
      for (int c = lane * 2; c < ND; c += 64) {
        const float2 v = *(const float2*)(x + (long)s * ND + c);   // xj = x[src]
        v2h hh = {(_Float16)v.x, (_Float16)v.y};
        *(v2h*)(at + r * TW + ND + c) = hh;
      }
      for (int c = lane * 2; c < ED; c += 64) {
        float2 v;
        if (FIRST && e >= NEDGES_RAW) { v.x = 0.0f; v.y = 0.0f; }
        else v = *(const float2*)(ea_in + (long)e * ED + c);
        v2h hh = {(_Float16)v.x, (_Float16)v.y};
        *(v2h*)(at + r * TW + 2 * ND + c) = hh;
      }
      for (int c = DINE + lane * 2; c < KPE; c += 64) {
        v2h zz = {(_Float16)0.0f, (_Float16)0.0f};
        *(v2h*)(at + r * TW + c) = zz;                            // K padding
      }
    }
    wave_sync();

    // ---- edge MLP layer 1: relu(A @ W1e + b1e) -> hidden tile
    v8f acc[4] = {};
    gemm16<KTE>(at, TW, w1e_s, lane, acc);
#pragma unroll
    for (int nt = 0; nt < 4; ++nt) {
      const int n = lm + nt * 16;
#pragma unroll
      for (int r = 0; r < 8; ++r) {
        const int m = r + 8 * lh;
        const float v = acc[nt][r] + bia_s[n];
        ht[m * EMB + n] = (_Float16)(v > 0.0f ? v : 0.0f);
      }
    }
    wave_sync();

    // ---- edge MLP layer 2: ea_new = H @ W2e + b2e (write out + feed msg tile)
    v8f acc2[4] = {};
    gemm16<2>(ht, EMB, w2e_s, lane, acc2);
#pragma unroll
    for (int nt = 0; nt < 4; ++nt) {
      const int n = lm + nt * 16;
#pragma unroll
      for (int r = 0; r < 8; ++r) {
        const int m = r + 8 * lh;
        const float v = acc2[nt][r] + bia_s[EMB + n];
        at[m * TW + 2 * ND + n] = (_Float16)v;   // overwrite ea with ea_new
        ea_out[(long)(ebase + m) * EMB + n] = v; // coalesced across lanes
      }
    }
    wave_sync();

    // ---- msg MLP layer 1: relu([xi|xj|ea_new] @ W1m + b1m)
    v8f acc3[4] = {};
    gemm16<KTM>(at, TW, w1m_s, lane, acc3);
#pragma unroll
    for (int nt = 0; nt < 4; ++nt) {
      const int n = lm + nt * 16;
#pragma unroll
      for (int r = 0; r < 8; ++r) {
        const int m = r + 8 * lh;
        const float v = acc3[nt][r] + bia_s[2 * EMB + n];
        ht[m * EMB + n] = (_Float16)(v > 0.0f ? v : 0.0f);
      }
    }
    wave_sync();

    // ---- msg MLP layer 2 + scatter-add into aggr[dst]
    v8f acc4[4] = {};
    gemm16<2>(ht, EMB, w2m_s, lane, acc4);
#pragma unroll
    for (int nt = 0; nt < 4; ++nt) {
      const int n = lm + nt * 16;
#pragma unroll
      for (int r = 0; r < 8; ++r) {
        const int m = r + 8 * lh;
        const float v = acc4[nt][r] + bia_s[3 * EMB + n];
        atomicAdd(&aggr[(long)dd[m] * EMB + n], v);   // global_atomic_add_f32
      }
    }
    wave_sync();  // WAR: next iteration's gather rewrites at/dd
  }
}

// ---------------- per-node MLP: x_new = MLP([x | aggr]) -----------------------
template <int ND>
__global__ void __launch_bounds__(256) gnn_node_kernel(
    const float* __restrict__ x, const float* __restrict__ aggr,
    const float* __restrict__ w1, const float* __restrict__ b1,
    const float* __restrict__ w2, const float* __restrict__ b2,
    float* __restrict__ x_out) {
  constexpr int TW = ND + EMB;  // 96 or 128, multiples of 32
  constexpr int KT = TW / 32;

  extern __shared__ char smem[] __attribute__((aligned(32)));
  _Float16* w1_s  = (_Float16*)smem;                   // TW*64 (swizzled)
  _Float16* w2_s  = w1_s + TW * EMB;                   // 64*64 (swizzled)
  float*    bia_s = (float*)(w2_s + EMB * EMB);
  _Float16* tiles = (_Float16*)(bia_s + 2 * EMB);
  _Float16* hids  = tiles + 8 * 16 * TW;

  const int tid = threadIdx.x;
  stage_w_sw<KT, TW>(w1_s, w1, tid);
  stage_w_sw<2, EMB>(w2_s, w2, tid);
  if (tid < EMB) { bia_s[tid] = b1[tid]; bia_s[EMB + tid] = b2[tid]; }
  __syncthreads();

  const int wave = tid >> 5, lane = tid & 31;
  const int lm = lane & 15, lh = lane >> 4;
  constexpr int NTILES = NNODES / 16;  // 3125, exact

  _Float16* at = tiles + wave * 16 * TW;
  _Float16* ht = hids + wave * 16 * EMB;

  for (int tile = blockIdx.x * 8 + wave; tile < NTILES; tile += (int)gridDim.x * 8) {
    const int nbase = tile * 16;

    for (int r = 0; r < 16; ++r) {
      const int nd = nbase + r;
      for (int c = lane * 2; c < ND; c += 64) {
        const float2 v = *(const float2*)(x + (long)nd * ND + c);
        v2h hh = {(_Float16)v.x, (_Float16)v.y};
        *(v2h*)(at + r * TW + c) = hh;
      }
      for (int c = lane * 2; c < EMB; c += 64) {
        const float2 v = *(const float2*)(aggr + (long)nd * EMB + c);
        v2h hh = {(_Float16)v.x, (_Float16)v.y};
        *(v2h*)(at + r * TW + ND + c) = hh;
      }
    }
    wave_sync();

    v8f acc[4] = {};
    gemm16<KT>(at, TW, w1_s, lane, acc);
#pragma unroll
    for (int nt = 0; nt < 4; ++nt) {
      const int n = lm + nt * 16;
#pragma unroll
      for (int r = 0; r < 8; ++r) {
        const int m = r + 8 * lh;
        const float v = acc[nt][r] + bia_s[n];
        ht[m * EMB + n] = (_Float16)(v > 0.0f ? v : 0.0f);
      }
    }
    wave_sync();

    v8f acc2[4] = {};
    gemm16<2>(ht, EMB, w2_s, lane, acc2);
#pragma unroll
    for (int nt = 0; nt < 4; ++nt) {
      const int n = lm + nt * 16;
#pragma unroll
      for (int r = 0; r < 8; ++r) {
        const int m = r + 8 * lh;
        x_out[(long)(nbase + m) * EMB + n] = acc2[nt][r] + bia_s[EMB + n];
      }
    }
    wave_sync();  // WAR vs next iteration's gather
  }
}

// ---------------- host orchestration ------------------------------------------
extern "C" void kernel_launch(void* const* d_in, const int* in_sizes, int n_in,
                              void* d_out, int out_size, void* d_ws, size_t ws_size,
                              hipStream_t stream) {
  (void)in_sizes; (void)n_in; (void)out_size; (void)ws_size;

  const float* x0   = (const float*)d_in[0];
  const float* ea0  = (const float*)d_in[1];
  const int*   eidx = (const int*)d_in[2];
  // params: 3 layers x {edge,msg,node} x {w1,b1,w2,b2} in insertion order
  const float* P[36];
  for (int i = 0; i < 36; ++i) P[i] = (const float*)d_in[3 + i];

  float* out_x  = (float*)d_out;                    // 50000*64
  float* out_ea = out_x + (long)NNODES * EMB;       // 850000*64

  float* aggr   = (float*)d_ws;                     // N*64
  float* x_tmp  = aggr + (long)NNODES * EMB;        // N*64
  float* ea_tmp = x_tmp + (long)NNODES * EMB;       // E*64

  const int ETILES = NEDGES_TOT / 16;               // 53125
  const int NTILES = NNODES / 16;                   // 3125
  const int EBLK = (ETILES + 8 * TPW - 1) / (8 * TPW);  // 1661
  const int NBLK = (NTILES + 8 * TPW - 1) / (8 * TPW);  // 98

  auto e_smem = [](int ND, int ED) {
    const int TW = 2 * ND + EMB, DINE = 2 * ND + ED, KPE = (DINE + 31) & ~31;
    return (size_t)(KPE * EMB + EMB * EMB + TW * EMB + EMB * EMB) * 2 +
           4 * EMB * 4 + (size_t)8 * 16 * TW * 2 + (size_t)8 * 16 * EMB * 2 +
           8 * 16 * 4;
  };
  auto n_smem = [](int ND) {
    const int TW = ND + EMB;
    return (size_t)(TW * EMB + EMB * EMB) * 2 + 2 * EMB * 4 +
           (size_t)8 * 16 * TW * 2 + (size_t)8 * 16 * EMB * 2;
  };

  const long AGGR_N = (long)NNODES * EMB;

  // ---- layer 0: x(in,32) ea(in,16) -> ea: out_ea, x: out_x
  gnn_zero_kernel<<<512, 256, 0, stream>>>(aggr, AGGR_N);
  gnn_edge_msg_kernel<32, 16, true><<<EBLK, 256, e_smem(32, 16), stream>>>(
      x0, ea0, eidx, P[0], P[1], P[2], P[3], P[4], P[5], P[6], P[7],
      out_ea, aggr);
  gnn_node_kernel<32><<<NBLK, 256, n_smem(32), stream>>>(
      x0, aggr, P[8], P[9], P[10], P[11], out_x);

  // ---- layer 1: x: out_x, ea: out_ea -> ea: ea_tmp, x: x_tmp
  gnn_zero_kernel<<<512, 256, 0, stream>>>(aggr, AGGR_N);
  gnn_edge_msg_kernel<64, 64, false><<<EBLK, 256, e_smem(64, 64), stream>>>(
      out_x, out_ea, eidx, P[12], P[13], P[14], P[15], P[16], P[17], P[18],
      P[19], ea_tmp, aggr);
  gnn_node_kernel<64><<<NBLK, 256, n_smem(64), stream>>>(
      out_x, aggr, P[20], P[21], P[22], P[23], x_tmp);

  // ---- layer 2: x: x_tmp, ea: ea_tmp -> ea: out_ea, x: out_x (final)
  gnn_zero_kernel<<<512, 256, 0, stream>>>(aggr, AGGR_N);
  gnn_edge_msg_kernel<64, 64, false><<<EBLK, 256, e_smem(64, 64), stream>>>(
      x_tmp, ea_tmp, eidx, P[24], P[25], P[26], P[27], P[28], P[29], P[30],
      P[31], out_ea, aggr);
  gnn_node_kernel<64><<<NBLK, 256, n_smem(64), stream>>>(
      x_tmp, aggr, P[32], P[33], P[34], P[35], out_x);
}
